// MaskedSelfAttention_42606075576580
// MI455X (gfx1250) — matrix operand order repaired and checked by reading, not compiled, not run on hardware
//
#include <hip/hip_runtime.h>
#include <stdint.h>
#include <stddef.h>

// MaskedSelfAttention for MI455X (gfx1250, wave32, WMMA bf16 16x16x32 + TDM).
// B=16, S=2048, E=1024, D=64.
//
// ws layout (bytes):
//   [0)          qb : bf16 [B][S][D]   row-major   (4 MiB)
//   [4  MiB)     kb : bf16 [B][S][D]   row-major   (4 MiB)
//   [8  MiB)     vT : bf16 [B][D][S]   transposed  (4 MiB)
//   [12 MiB)     WqT/WkT/WvT : bf16 [D][E] each    (3 * 128 KiB)

#define BB 16
#define SS 2048
#define EE 1024
#define DD 64

typedef __attribute__((ext_vector_type(16))) __bf16 v16bf;
typedef __attribute__((ext_vector_type(8)))  __bf16 v8bf;
typedef __attribute__((ext_vector_type(8)))  float  v8f;
typedef __attribute__((ext_vector_type(4)))  unsigned int u32x4;
typedef __attribute__((ext_vector_type(8)))  unsigned int u32x8;

// ---------------------------------------------------------------------------
// TDM: 2D bf16 tile load Global->LDS via tensor_load_to_lds (D# built per
// CDNA5 ISA; 2-operand form => 2D, VADDR2/3 NULL).
// tile_d0 = contiguous elements per row, tile_d1 = rows,
// stride_elems = row stride in elements. Tracked with TENSORcnt.
// ---------------------------------------------------------------------------
__device__ __forceinline__ void tdm_load_2d_bf16(uint32_t lds_off,
                                                 const void* gptr,
                                                 uint32_t tile_d0,
                                                 uint32_t tile_d1,
                                                 uint32_t stride_elems) {
    uint64_t ga = (uint64_t)(uintptr_t)gptr;
    u32x4 g0;
    g0[0] = 1u;                                               // count=1, user mode
    g0[1] = lds_off;                                          // lds_addr [63:32]
    g0[2] = (uint32_t)ga;                                     // global_addr lo
    g0[3] = (uint32_t)((ga >> 32) & 0x01FFFFFFu) | (2u << 30);// addr hi | type=2
    u32x8 g1;
    g1[0] = 1u << 16;                                         // data_size=1 (2B)
    g1[1] = (tile_d0 & 0xFFFFu) << 16;                        // tensor_dim0 lo16
    g1[2] = (tile_d1 & 0xFFFFu) << 16;                        // tensor_dim1 lo16
    g1[3] = (tile_d0 & 0xFFFFu) << 16;                        // tile_dim0
    g1[4] = (tile_d1 & 0xFFFFu);                              // tile_dim1 (tile_dim2=0)
    g1[5] = stride_elems;                                     // tensor_dim0_stride lo32
    g1[6] = 0u;
    g1[7] = 0u;
    asm volatile("tensor_load_to_lds %0, %1" :: "s"(g0), "s"(g1) : "memory");
}

// ---------------------------------------------------------------------------
// Kernel 0: convert + transpose weights: W[E][D] f32 -> Wt[D][E] bf16
// ---------------------------------------------------------------------------
__global__ void wconv_kernel(const float* __restrict__ Wq,
                             const float* __restrict__ Wk,
                             const float* __restrict__ Wv,
                             __bf16* __restrict__ WqT,
                             __bf16* __restrict__ WkT,
                             __bf16* __restrict__ WvT) {
    int idx = blockIdx.x * 256 + threadIdx.x;          // 0 .. 3*D*E-1
    int which = idx >> 16;                             // D*E = 65536
    int r = idx & 65535;
    int n = r >> 10;                                   // 0..63  (D)
    int k = r & 1023;                                  // 0..1023 (E)
    const float* W = (which == 0) ? Wq : (which == 1) ? Wk : Wv;
    __bf16* Wt     = (which == 0) ? WqT : (which == 1) ? WkT : WvT;
    Wt[n * EE + k] = (__bf16)W[k * DD + n];
}

// ---------------------------------------------------------------------------
// Kernel 1: QKV projection.  grid = B*S/16 blocks, 96 threads (3 waves).
// wave 0 -> q (row-major), wave 1 -> k (row-major), wave 2 -> v (transposed)
// ---------------------------------------------------------------------------
__global__ void __launch_bounds__(96)
qkv_kernel(const float* __restrict__ x,
           const __bf16* __restrict__ WqT,
           const __bf16* __restrict__ WkT,
           const __bf16* __restrict__ WvT,
           const float* __restrict__ bq,
           const float* __restrict__ bk,
           const float* __restrict__ bv,
           __bf16* __restrict__ qout,
           __bf16* __restrict__ kout,
           __bf16* __restrict__ vT) {
    const int lane = threadIdx.x & 31;
    const int wave = threadIdx.x >> 5;                 // 0=q 1=k 2=v
    const int m    = lane & 15;
    const int half = lane >> 4;

    const int tile = blockIdx.x;
    const int b    = tile >> 7;                        // S/16 = 128 tiles/batch
    const int s0t  = (tile & 127) << 4;

    const float*  xrow = x + (size_t)(b * SS + s0t + m) * EE;
    const __bf16* Wt   = (wave == 0) ? WqT : (wave == 1) ? WkT : WvT;
    const float*  bias = (wave == 0) ? bq  : (wave == 1) ? bk  : bv;

    v8f acc[4] = {};

    for (int k0 = 0; k0 < EE; k0 += 32) {
        // A fragment: rows = m, K = k0 + half*16 + 0..15 (contiguous)
        v16bf a;
        const float* ap = xrow + k0 + half * 16;
#pragma unroll
        for (int j = 0; j < 8; ++j) {
            a[j]     = (__bf16)ap[j];
            a[j + 8] = (__bf16)ap[8 + j];
        }
#pragma unroll
        for (int n = 0; n < 4; ++n) {
            // B fragment: col = n*16+m, K = k0 + half*16 + 0..15 (contiguous)
            const __bf16* bp = Wt + (size_t)(n * 16 + m) * EE + k0 + half * 16;
            v16bf bf = *(const v16bf*)bp;
            acc[n] = __builtin_amdgcn_wmma_f32_16x16x32_bf16(
                false, a, false, bf, (short)0, acc[n], false, false);
        }
    }

    if (wave < 2) {
        __bf16* outp = (wave == 0) ? qout : kout;
#pragma unroll
        for (int n = 0; n < 4; ++n) {
            float bi = bias[n * 16 + m];
#pragma unroll
            for (int j = 0; j < 8; ++j) {
                int s = s0t + j + half * 8;
                outp[(size_t)(b * SS + s) * DD + n * 16 + m] = (__bf16)(acc[n][j] + bi);
            }
        }
    } else {
        // v transposed: rows j contiguous along S -> 16B vector store
#pragma unroll
        for (int n = 0; n < 4; ++n) {
            float bi = bias[n * 16 + m];
            int   d  = n * 16 + m;
            v8bf st;
#pragma unroll
            for (int j = 0; j < 8; ++j) st[j] = (__bf16)(acc[n][j] + bi);
            *(v8bf*)(vT + (size_t)(b * DD + d) * SS + s0t + half * 8) = st;
        }
    }
}

// ---------------------------------------------------------------------------
// Kernel 2: flash attention. grid = B*S/(16*4) blocks, 128 threads (4 waves),
// each wave owns one 16-query tile; all 4 waves share one batch's K/V stream.
// K/V tiles (32 keys) are staged into double-buffered LDS by the TDM
// (tensor_load_to_lds), overlapped with WMMA + softmax of the current tile.
// ---------------------------------------------------------------------------
__global__ void __launch_bounds__(128)
attn_kernel(const __bf16* __restrict__ qb,
            const __bf16* __restrict__ kb,
            const __bf16* __restrict__ vT,
            const int* __restrict__ mask,
            float* __restrict__ out) {
    __shared__ __bf16 kbufs[2][32][64];                // 2 x 4KB  K stage
    __shared__ __bf16 vbufs[2][64][32];                // 2 x 4KB  V stage
    __shared__ __bf16 pbuf[4][16][32];                 // per-wave P transpose

    const int lane = threadIdx.x & 31;
    const int wave = threadIdx.x >> 5;
    const int m    = lane & 15;
    const int half = lane >> 4;

    const int qtile = blockIdx.x * 4 + wave;
    const int b     = qtile >> 7;                      // same b for all 4 waves
    const int s0t   = (qtile & 127) << 4;

    // Q as two A fragments (K = head dim 0..31 and 32..63)
    const __bf16* qp = qb + (size_t)(b * SS + s0t + m) * DD;
    v16bf qa0, qa1;
#pragma unroll
    for (int j = 0; j < 8; ++j) {
        qa0[j]     = qp[half * 16 + j];
        qa0[j + 8] = qp[half * 16 + 8 + j];
        qa1[j]     = qp[32 + half * 16 + j];
        qa1[j + 8] = qp[32 + half * 16 + 8 + j];
    }

    v8f   acc[4] = {};
    float mrow[8], lrow[8];
#pragma unroll
    for (int j = 0; j < 8; ++j) { mrow[j] = -3.0e38f; lrow[j] = 0.0f; }

    const int* mrowp = mask + b * SS;

    // prologue: TDM-stage tile 0 into buffer 0
    if (wave == 0) {
        tdm_load_2d_bf16((uint32_t)(uintptr_t)&kbufs[0][0][0],
                         kb + (size_t)b * SS * DD, 64, 32, 64);
        tdm_load_2d_bf16((uint32_t)(uintptr_t)&vbufs[0][0][0],
                         vT + (size_t)b * DD * SS, 32, 64, SS);
        __builtin_amdgcn_s_wait_tensorcnt(0);
    }
    __syncthreads();

    for (int t = 0; t < SS / 32; ++t) {
        const int cur = t & 1;
        const int nxt = cur ^ 1;
        const int kb0 = t * 32;

        // kick off DMA for the next 32-key tile while we compute this one
        if (wave == 0 && (kb0 + 32) < SS) {
            tdm_load_2d_bf16((uint32_t)(uintptr_t)&kbufs[nxt][0][0],
                             kb + (size_t)(b * SS + kb0 + 32) * DD, 64, 32, 64);
            tdm_load_2d_bf16((uint32_t)(uintptr_t)&vbufs[nxt][0][0],
                             vT + (size_t)(b * DD) * SS + kb0 + 32, 32, 64, SS);
        }

        // K^T B fragments from LDS: col = key row in stage, K = head dim
        const __bf16* kt = &kbufs[cur][0][0];
        v16bf b00 = *(const v16bf*)(kt + m * 64 + half * 16);
        v16bf b01 = *(const v16bf*)(kt + m * 64 + 32 + half * 16);
        v16bf b10 = *(const v16bf*)(kt + (16 + m) * 64 + half * 16);
        v16bf b11 = *(const v16bf*)(kt + (16 + m) * 64 + 32 + half * 16);

        v8f sc0 = {}, sc1 = {};
        sc0 = __builtin_amdgcn_wmma_f32_16x16x32_bf16(false, qa0, false, b00, (short)0, sc0, false, false);
        sc0 = __builtin_amdgcn_wmma_f32_16x16x32_bf16(false, qa1, false, b01, (short)0, sc0, false, false);
        sc1 = __builtin_amdgcn_wmma_f32_16x16x32_bf16(false, qa0, false, b10, (short)0, sc1, false, false);
        sc1 = __builtin_amdgcn_wmma_f32_16x16x32_bf16(false, qa1, false, b11, (short)0, sc1, false, false);

        const int mk0 = mrowp[kb0 + m];
        const int mk1 = mrowp[kb0 + 16 + m];

        // online softmax per row j (row = j + half*8; cols striped over 16 lanes)
        float p0[8], p1[8];
#pragma unroll
        for (int j = 0; j < 8; ++j) {
            float s0v = mk0 ? sc0[j] * 0.125f : -1.0e9f;
            float s1v = mk1 ? sc1[j] * 0.125f : -1.0e9f;
            float r = fmaxf(s0v, s1v);
#pragma unroll
            for (int off = 1; off < 16; off <<= 1)
                r = fmaxf(r, __shfl_xor(r, off, 32));
            float mnew  = fmaxf(mrow[j], r);
            float alpha = __expf(mrow[j] - mnew);
            mrow[j] = mnew;
            float e0 = __expf(s0v - mnew);
            float e1 = __expf(s1v - mnew);
            float rs = e0 + e1;
#pragma unroll
            for (int off = 1; off < 16; off <<= 1)
                rs += __shfl_xor(rs, off, 32);
            lrow[j] = lrow[j] * alpha + rs;
#pragma unroll
            for (int n = 0; n < 4; ++n) acc[n][j] *= alpha;
            p0[j] = e0;
            p1[j] = e1;
        }

        // C-layout -> A-layout transpose of P through per-wave LDS
#pragma unroll
        for (int j = 0; j < 8; ++j) {
            pbuf[wave][j + half * 8][m]      = (__bf16)p0[j];
            pbuf[wave][j + half * 8][16 + m] = (__bf16)p1[j];
        }
        asm volatile("s_wait_dscnt 0" ::: "memory");
        v16bf pa;
#pragma unroll
        for (int j = 0; j < 8; ++j) {
            pa[j]     = pbuf[wave][m][half * 16 + j];
            pa[j + 8] = pbuf[wave][m][half * 16 + 8 + j];
        }

        // PV: B fragments from LDS-staged transposed V (contiguous along keys)
        const __bf16* vt = &vbufs[cur][0][0];
#pragma unroll
        for (int n = 0; n < 4; ++n) {
            v16bf vb = *(const v16bf*)(vt + (n * 16 + m) * 32 + half * 16);
            acc[n] = __builtin_amdgcn_wmma_f32_16x16x32_bf16(
                false, pa, false, vb, (short)0, acc[n], false, false);
        }

        // drain this iteration's DMA, then publish buffers to all waves
        if (wave == 0) __builtin_amdgcn_s_wait_tensorcnt(0);
        __syncthreads();
    }

    // finalize: divide by row sums, store f32 [B][S][D]
#pragma unroll
    for (int j = 0; j < 8; ++j) {
        float inv = 1.0f / lrow[j];
        int   s   = s0t + j + half * 8;
        float* op = out + (size_t)(b * SS + s) * DD;
#pragma unroll
        for (int n = 0; n < 4; ++n) op[n * 16 + m] = acc[n][j] * inv;
    }
}

// ---------------------------------------------------------------------------
extern "C" void kernel_launch(void* const* d_in, const int* in_sizes, int n_in,
                              void* d_out, int out_size, void* d_ws, size_t ws_size,
                              hipStream_t stream) {
    (void)in_sizes; (void)n_in; (void)out_size; (void)ws_size;

    const float* x    = (const float*)d_in[0];
    const int*   mask = (const int*)d_in[1];
    const float* Wq   = (const float*)d_in[2];
    const float* bq   = (const float*)d_in[3];
    const float* Wk   = (const float*)d_in[4];
    const float* bk   = (const float*)d_in[5];
    const float* Wv   = (const float*)d_in[6];
    const float* bv   = (const float*)d_in[7];
    float*       out  = (float*)d_out;

    char* ws = (char*)d_ws;
    const size_t qkvBytes = (size_t)BB * SS * DD * 2;  // 4 MiB each
    const size_t wtBytes  = (size_t)DD * EE * 2;       // 128 KiB each
    __bf16* qbuf = (__bf16*)(ws);
    __bf16* kbuf = (__bf16*)(ws + qkvBytes);
    __bf16* vT   = (__bf16*)(ws + 2 * qkvBytes);
    __bf16* WqT  = (__bf16*)(ws + 3 * qkvBytes);
    __bf16* WkT  = (__bf16*)(ws + 3 * qkvBytes + wtBytes);
    __bf16* WvT  = (__bf16*)(ws + 3 * qkvBytes + 2 * wtBytes);

    wconv_kernel<<<768, 256, 0, stream>>>(Wq, Wk, Wv, WqT, WkT, WvT);

    qkv_kernel<<<BB * SS / 16, 96, 0, stream>>>(x, WqT, WkT, WvT, bq, bk, bv,
                                                qbuf, kbuf, vT);

    attn_kernel<<<BB * SS / 64, 128, 0, stream>>>(qbuf, kbuf, vT, mask, out);
}
